// GINELayer_88845693485603
// MI455X (gfx1250) — matrix operand order repaired and checked by reading, not compiled, not run on hardware
//
#include <hip/hip_runtime.h>
#include <hip/hip_bf16.h>
#include <stdint.h>

// ---------------------------------------------------------------------------
// GINE layer for MI455X (gfx1250, wave32, WMMA).
// Precision: bf16 A/B fragments, fp32 WMMA accumulation
// (v_wmma_f32_16x16x32_bf16). bf16 keeps the 168-GFLOP edge MLP on the
// memory roofline (~60us @ 23.3 TB/s); fp32 WMMA (16x16x4) would be ~8x
// slower per instruction and make the layer compute-bound.
// ---------------------------------------------------------------------------

typedef __attribute__((ext_vector_type(16))) __bf16 v16bf;
typedef __attribute__((ext_vector_type(8)))  float  v8f;

#define GN_NODES 20000
#define GN_EDGES 320000
#define GN_INF   256
#define GN_HID   512
#define GN_OUTF  256

#if defined(__has_builtin)
#if __has_builtin(__builtin_amdgcn_cvt_pk_bf16_f32)
#define GN_HAVE_PK_BF16 1
#endif
#endif

union BfFrag {
    v16bf    v;
    uint32_t u[8];
};

__device__ __forceinline__ uint32_t f2bf_bits(float f) {
    union { float f; uint32_t u; } c;
    c.f = f;
    uint32_t u = c.u;
    // round-to-nearest-even bf16 truncation
    return (u + 0x7FFFu + ((u >> 16) & 1u)) >> 16;
}

// Pack two fp32 -> packed bf16x2 in one VALU op when the HW cvt is exposed.
__device__ __forceinline__ uint32_t pack2bf(float lo, float hi) {
#ifdef GN_HAVE_PK_BF16
    typedef __attribute__((ext_vector_type(2))) __bf16 v2bf;
    union { v2bf v; uint32_t u; } c;
    c.v = __builtin_amdgcn_cvt_pk_bf16_f32(lo, hi);
    return c.u;
#else
    return f2bf_bits(lo) | (f2bf_bits(hi) << 16);
#endif
}

// A-matrix 16x32 bf16 lane layout (ISA 7.12.2):
// lanes 0-15 : M=lane,   VGPR0-3 = K[kOff+0..7],  VGPR4-7 = K[kOff+16..23]
// lanes 16-31: M=lane-16,VGPR0-3 = K[kOff+8..15], VGPR4-7 = K[kOff+24..31]
__device__ __forceinline__ BfFrag load_a_f32row(const float* __restrict__ row,
                                                int kOff, int half) {
    BfFrag f;
    const float4* p1 = reinterpret_cast<const float4*>(row + kOff + half * 8);
    const float4* p2 = reinterpret_cast<const float4*>(row + kOff + 16 + half * 8);
    float4 a0 = p1[0], a1 = p1[1];
    float4 b0 = p2[0], b1 = p2[1];
    f.u[0] = pack2bf(a0.x, a0.y); f.u[1] = pack2bf(a0.z, a0.w);
    f.u[2] = pack2bf(a1.x, a1.y); f.u[3] = pack2bf(a1.z, a1.w);
    f.u[4] = pack2bf(b0.x, b0.y); f.u[5] = pack2bf(b0.z, b0.w);
    f.u[6] = pack2bf(b1.x, b1.y); f.u[7] = pack2bf(b1.z, b1.w);
    return f;
}

// Fused (1+eps)*node + out_h A-frag loader (node kernel stage 1).
__device__ __forceinline__ BfFrag load_a_combine(const float* __restrict__ nrow,
                                                 const float* __restrict__ hrow,
                                                 float s, int kOff, int half) {
    BfFrag f;
    const float4* pn1 = reinterpret_cast<const float4*>(nrow + kOff + half * 8);
    const float4* pn2 = reinterpret_cast<const float4*>(nrow + kOff + 16 + half * 8);
    const float4* ph1 = reinterpret_cast<const float4*>(hrow + kOff + half * 8);
    const float4* ph2 = reinterpret_cast<const float4*>(hrow + kOff + 16 + half * 8);
    float4 n0 = pn1[0], n1 = pn1[1], n2 = pn2[0], n3 = pn2[1];
    float4 h0 = ph1[0], h1 = ph1[1], h2 = ph2[0], h3 = ph2[1];
    f.u[0] = pack2bf(s * n0.x + h0.x, s * n0.y + h0.y);
    f.u[1] = pack2bf(s * n0.z + h0.z, s * n0.w + h0.w);
    f.u[2] = pack2bf(s * n1.x + h1.x, s * n1.y + h1.y);
    f.u[3] = pack2bf(s * n1.z + h1.z, s * n1.w + h1.w);
    f.u[4] = pack2bf(s * n2.x + h2.x, s * n2.y + h2.y);
    f.u[5] = pack2bf(s * n2.z + h2.z, s * n2.w + h2.w);
    f.u[6] = pack2bf(s * n3.x + h3.x, s * n3.y + h3.y);
    f.u[7] = pack2bf(s * n3.z + h3.z, s * n3.w + h3.w);
    return f;
}

// A-frag from bf16 LDS row (already packed pairs).
__device__ __forceinline__ BfFrag load_a_lds(const uint16_t* row, int kOff, int half) {
    BfFrag f;
    const uint4* p1 = reinterpret_cast<const uint4*>(row + kOff + half * 8);
    const uint4* p2 = reinterpret_cast<const uint4*>(row + kOff + 16 + half * 8);
    uint4 q0 = *p1, q1 = *p2;
    f.u[0] = q0.x; f.u[1] = q0.y; f.u[2] = q0.z; f.u[3] = q0.w;
    f.u[4] = q1.x; f.u[5] = q1.y; f.u[6] = q1.z; f.u[7] = q1.w;
    return f;
}

// B-matrix 32x16 bf16: lane holds column n=lane&15; lanes 0-15 K=kOff+0..7,16..23,
// lanes 16-31 K=kOff+8..15,24..31, packed 2/VGPR.  Weights stored transposed
// [N][K] bf16 so each lane loads 2x16 contiguous bytes (L2-resident).
__device__ __forceinline__ BfFrag load_b(const uint16_t* __restrict__ colRow,
                                         int kOff, int half) {
    BfFrag f;
    const uint4* p1 = reinterpret_cast<const uint4*>(colRow + kOff + half * 8);
    const uint4* p2 = reinterpret_cast<const uint4*>(colRow + kOff + 16 + half * 8);
    uint4 q0 = *p1, q1 = *p2;
    f.u[0] = q0.x; f.u[1] = q0.y; f.u[2] = q0.z; f.u[3] = q0.w;
    f.u[4] = q1.x; f.u[5] = q1.y; f.u[6] = q1.z; f.u[7] = q1.w;
    return f;
}

__device__ __forceinline__ v8f wmma_bf16(const BfFrag& a, const BfFrag& b, v8f c) {
    return __builtin_amdgcn_wmma_f32_16x16x32_bf16(false, a.v, false, b.v,
                                                   (short)0, c, false, false);
}

// ---------------------------------------------------------------------------
// Prep kernels
// ---------------------------------------------------------------------------
__global__ void cvt_transpose_kernel(const float* __restrict__ W,
                                     uint16_t* __restrict__ Wt, int K, int N) {
    int idx = blockIdx.x * 256 + threadIdx.x;
    if (idx < K * N) {
        int k = idx / N, n = idx % N;
        Wt[(size_t)n * K + k] = (uint16_t)f2bf_bits(W[idx]);
    }
}

__global__ void zero_f32_kernel(float* __restrict__ p, int n) {
    int i = blockIdx.x * 256 + threadIdx.x;
    if (i < n) p[i] = 0.0f;
}

// ---------------------------------------------------------------------------
// Edge kernel: e = MLP2(edge_feat); msg = relu(node[src]+e); atomic-add to h[dst]
// Block: 256 threads (8 waves), 128 edge rows per block (E = 2500 * 128 exact)
// ---------------------------------------------------------------------------
__global__ __launch_bounds__(256) void gine_edge_kernel(
    const float* __restrict__ edge_feat, const float* __restrict__ node_feat,
    const int* __restrict__ src, const int* __restrict__ dst,
    const uint16_t* __restrict__ W1t, const float* __restrict__ b1,
    const uint16_t* __restrict__ W2t, const float* __restrict__ b2,
    float* __restrict__ out_h) {
    __shared__ uint16_t sH[128 * GN_HID];          // 128 KB bf16 hidden tile

    const int lane   = threadIdx.x & 31;
    const int wave   = threadIdx.x >> 5;           // 0..7 -> 16-row strip
    const int half   = lane >> 4;
    const int l15    = lane & 15;
    const int blockM = blockIdx.x * 128;

    // ---- stage 1: X[16,256] @ W1t -> H[16,512], relu, -> LDS (per wave) ----
    BfFrag AF[8];
    {
        const float* rp = edge_feat + (size_t)(blockM + wave * 16 + l15) * GN_INF;
#pragma unroll
        for (int k = 0; k < 8; ++k) AF[k] = load_a_f32row(rp, k * 32, half);
    }
    for (int nt = 0; nt < 32; ++nt) {
        v8f acc = {};
        const uint16_t* wcol = W1t + (size_t)(nt * 16 + l15) * GN_INF;
#pragma unroll
        for (int k = 0; k < 8; ++k) {
            BfFrag BF = load_b(wcol, k * 32, half);
            acc = wmma_bf16(AF[k], BF, acc);
        }
        const int col  = nt * 16 + l15;
        const float bi = b1[col];
#pragma unroll
        for (int r = 0; r < 8; r += 2) {           // D layout: M = r + 8*half
            int m = wave * 16 + r + half * 8;
            uint32_t p = pack2bf(fmaxf(acc[r] + bi, 0.0f),
                                 fmaxf(acc[r + 1] + bi, 0.0f));
            sH[m * GN_HID + col]       = (uint16_t)(p & 0xFFFFu);
            sH[(m + 1) * GN_HID + col] = (uint16_t)(p >> 16);
        }
    }
    __syncthreads();

    // ---- stage 2: H[16,512] @ W2t -> Y[16,256]; fused gather/relu/scatter ----
    const uint16_t* hrow = sH + (size_t)(wave * 16 + l15) * GN_HID;
    for (int nt = 0; nt < 16; ++nt) {
        v8f acc = {};
        const uint16_t* wcol = W2t + (size_t)(nt * 16 + l15) * GN_HID;
#pragma unroll
        for (int k = 0; k < 16; ++k) {
            BfFrag A2 = load_a_lds(hrow, k * 32, half);
            BfFrag BF = load_b(wcol, k * 32, half);
            acc = wmma_bf16(A2, BF, acc);
        }
        const int col  = nt * 16 + l15;
        const float bi = b2[col];
#pragma unroll
        for (int r = 0; r < 8; ++r) {
            int m = blockM + wave * 16 + r + half * 8;
            if (m < GN_EDGES) {
                int s = src[m], d = dst[m];
                float msg = node_feat[(size_t)s * GN_INF + col] + acc[r] + bi;
                msg = fmaxf(msg, 0.0f);
                __hip_atomic_fetch_add(&out_h[(size_t)d * GN_INF + col], msg,
                                       __ATOMIC_RELAXED, __HIP_MEMORY_SCOPE_AGENT);
            }
        }
    }
}

// ---------------------------------------------------------------------------
// Node kernel: out = MLP2((1+eps)*node_feat + out_h)
// ---------------------------------------------------------------------------
__global__ __launch_bounds__(256) void gine_node_kernel(
    const float* __restrict__ node_feat, const float* __restrict__ out_h,
    const uint16_t* __restrict__ W1t, const float* __restrict__ b1,
    const uint16_t* __restrict__ W2t, const float* __restrict__ b2,
    const float* __restrict__ eps, float* __restrict__ out) {
    __shared__ uint16_t sH[128 * GN_HID];

    const int lane   = threadIdx.x & 31;
    const int wave   = threadIdx.x >> 5;
    const int half   = lane >> 4;
    const int l15    = lane & 15;
    const int blockM = blockIdx.x * 128;
    const float scl  = 1.0f + eps[0];

    // ---- stage 1 ----
    BfFrag AF[8];
    {
        int row = blockM + wave * 16 + l15;
        if (row >= GN_NODES) row = GN_NODES - 1;   // clamp; stores are masked
        const float* nr = node_feat + (size_t)row * GN_INF;
        const float* hr = out_h + (size_t)row * GN_INF;
#pragma unroll
        for (int k = 0; k < 8; ++k) AF[k] = load_a_combine(nr, hr, scl, k * 32, half);
    }
    for (int nt = 0; nt < 32; ++nt) {
        v8f acc = {};
        const uint16_t* wcol = W1t + (size_t)(nt * 16 + l15) * GN_INF;
#pragma unroll
        for (int k = 0; k < 8; ++k) {
            BfFrag BF = load_b(wcol, k * 32, half);
            acc = wmma_bf16(AF[k], BF, acc);
        }
        const int col  = nt * 16 + l15;
        const float bi = b1[col];
#pragma unroll
        for (int r = 0; r < 8; r += 2) {
            int m = wave * 16 + r + half * 8;
            uint32_t p = pack2bf(fmaxf(acc[r] + bi, 0.0f),
                                 fmaxf(acc[r + 1] + bi, 0.0f));
            sH[m * GN_HID + col]       = (uint16_t)(p & 0xFFFFu);
            sH[(m + 1) * GN_HID + col] = (uint16_t)(p >> 16);
        }
    }
    __syncthreads();

    // ---- stage 2 ----
    const uint16_t* hrow = sH + (size_t)(wave * 16 + l15) * GN_HID;
    for (int nt = 0; nt < 16; ++nt) {
        v8f acc = {};
        const uint16_t* wcol = W2t + (size_t)(nt * 16 + l15) * GN_HID;
#pragma unroll
        for (int k = 0; k < 16; ++k) {
            BfFrag A2 = load_a_lds(hrow, k * 32, half);
            BfFrag BF = load_b(wcol, k * 32, half);
            acc = wmma_bf16(A2, BF, acc);
        }
        const int col  = nt * 16 + l15;
        const float bi = b2[col];
#pragma unroll
        for (int r = 0; r < 8; ++r) {
            int m = blockM + wave * 16 + r + half * 8;
            if (m < GN_NODES) out[(size_t)m * GN_OUTF + col] = acc[r] + bi;
        }
    }
}

// ---------------------------------------------------------------------------
// Launch
// ---------------------------------------------------------------------------
extern "C" void kernel_launch(void* const* d_in, const int* in_sizes, int n_in,
                              void* d_out, int out_size, void* d_ws, size_t ws_size,
                              hipStream_t stream) {
    const float* node_feat = (const float*)d_in[0];
    const float* edge_feat = (const float*)d_in[1];
    const int*   src       = (const int*)d_in[2];
    const int*   dst       = (const int*)d_in[3];
    const float* We1       = (const float*)d_in[4];
    const float* be1       = (const float*)d_in[5];
    const float* We2       = (const float*)d_in[6];
    const float* be2       = (const float*)d_in[7];
    const float* Wn1       = (const float*)d_in[8];
    const float* bn1       = (const float*)d_in[9];
    const float* Wn2       = (const float*)d_in[10];
    const float* bn2       = (const float*)d_in[11];
    const float* eps       = (const float*)d_in[12];

    // Workspace layout (needs ~21.5 MB):
    //   [0, 256K)        We1^T bf16 [512][256]
    //   [256K, 512K)     We2^T bf16 [256][512]
    //   [512K, 768K)     Wn1^T bf16 [512][256]
    //   [768K, 1M)       Wn2^T bf16 [256][512]
    //   [1M, 1M+20.5M)   out_h fp32 [20000][256]
    uint8_t*  ws    = (uint8_t*)d_ws;
    uint16_t* W1t   = (uint16_t*)(ws + 0);
    uint16_t* W2t   = (uint16_t*)(ws + (256u << 10));
    uint16_t* Wn1t  = (uint16_t*)(ws + (512u << 10));
    uint16_t* Wn2t  = (uint16_t*)(ws + (768u << 10));
    float*    out_h = (float*)(ws + (1024u << 10));

    const int wElems = GN_INF * GN_HID;            // 131072 per matrix
    cvt_transpose_kernel<<<(wElems + 255) / 256, 256, 0, stream>>>(We1, W1t, GN_INF, GN_HID);
    cvt_transpose_kernel<<<(wElems + 255) / 256, 256, 0, stream>>>(We2, W2t, GN_HID, GN_INF);
    cvt_transpose_kernel<<<(wElems + 255) / 256, 256, 0, stream>>>(Wn1, Wn1t, GN_INF, GN_HID);
    cvt_transpose_kernel<<<(wElems + 255) / 256, 256, 0, stream>>>(Wn2, Wn2t, GN_HID, GN_INF);

    const int hElems = GN_NODES * GN_INF;
    zero_f32_kernel<<<(hElems + 255) / 256, 256, 0, stream>>>(out_h, hElems);

    gine_edge_kernel<<<GN_EDGES / 128, 256, 0, stream>>>(
        edge_feat, node_feat, src, dst, W1t, be1, W2t, be2, out_h);

    gine_node_kernel<<<(GN_NODES + 127) / 128, 256, 0, stream>>>(
        node_feat, out_h, Wn1t, bn1, Wn2t, bn2, eps, (float*)d_out);
}
